// MultiHeadAttention_66236985639126
// MI455X (gfx1250) — matrix operand order repaired and checked by reading, not compiled, not run on hardware
//
#include <hip/hip_runtime.h>

#define NHEADS 16
#define HIDDEN 1024
#define DK     64
#define BATCH  2
#define SEQ    2048
#define MROWS  (BATCH * SEQ)   // 4096

typedef _Float16 v16h __attribute__((ext_vector_type(16)));
typedef _Float16 v8h  __attribute__((ext_vector_type(8)));
typedef float    v8f  __attribute__((ext_vector_type(8)));

union HFrag {
    v16h v;
    v8h  seg[2];
    _Float16 h[16];
};

__device__ __forceinline__ v8f wmma16(const HFrag& a, const HFrag& b, v8f c) {
    // D = A(16x32, f16) x B(32x16, f16) + C(16x16, f32)
    return __builtin_amdgcn_wmma_f32_16x16x32_f16(
        /*neg_a=*/false, a.v, /*neg_b=*/false, b.v,
        /*c_mod=*/(short)0, c, /*reuse_a=*/false, /*reuse_b=*/false);
}

// Output layout modes for the GEMM kernel
#define OUT_HS    0   // f16, head-split [B, NH, SEQ, DK]
#define OUT_HS_T  1   // f16, head-split transposed [B, NH, DK, SEQ]   (for V)
#define OUT_F32   2   // f32, linear [M, N]  (final output + bias)

// ---------------------------------------------------------------------------
// GEMM: Y[M,N] = X[M,K] @ W[N,K]^T + bias[N]      (M=4096, N=K=1024)
// Block: 256 threads = 8 waves; wave tile 32x32 (2x2 WMMA); block tile 64x128.
// Operand set (X fp32 16MB + W 4MB) is L2-resident (192 MB) -> no LDS staging.
// ---------------------------------------------------------------------------
template <bool IN_F16, int MODE>
__global__ void __launch_bounds__(256)
gemm_wmma_kernel(const void* __restrict__ Xin,
                 const float* __restrict__ W,
                 const float* __restrict__ bias,
                 void* __restrict__ Yout)
{
    constexpr int K = HIDDEN;
    constexpr int N = HIDDEN;

    const int tid  = threadIdx.x;
    const int wave = tid >> 5;
    const int lane = tid & 31;
    const int half = lane >> 4;   // 0: lanes 0-15, 1: lanes 16-31
    const int l    = lane & 15;

    const int rowTop0 = blockIdx.y * 64  + (wave >> 2) * 32;  // + mi*16
    const int colTop0 = blockIdx.x * 128 + (wave & 3)  * 32;  // + ni*16

    const float*    Xf = (const float*)Xin;
    const _Float16* Xh = (const _Float16*)Xin;

    v8f acc[2][2] = {};

    for (int k0 = 0; k0 < K; k0 += 32) {
        // ---- A fragments (16x32 MxK): lane holds row (rowTop+l),
        //      K runs [kb1, kb1+8) and [kb1+16, kb1+24), kb1 = half*8
        HFrag a[2];
        const int kb1 = k0 + half * 8;
        #pragma unroll
        for (int mi = 0; mi < 2; ++mi) {
            const size_t base = (size_t)(rowTop0 + mi * 16 + l) * K + kb1;
            if constexpr (IN_F16) {
                a[mi].seg[0] = *(const v8h*)(Xh + base);
                a[mi].seg[1] = *(const v8h*)(Xh + base + 16);
            } else {
                const float4 f0 = ((const float4*)(Xf + base))[0];
                const float4 f1 = ((const float4*)(Xf + base))[1];
                const float4 f2 = ((const float4*)(Xf + base + 16))[0];
                const float4 f3 = ((const float4*)(Xf + base + 16))[1];
                a[mi].h[0]=(_Float16)f0.x; a[mi].h[1]=(_Float16)f0.y;
                a[mi].h[2]=(_Float16)f0.z; a[mi].h[3]=(_Float16)f0.w;
                a[mi].h[4]=(_Float16)f1.x; a[mi].h[5]=(_Float16)f1.y;
                a[mi].h[6]=(_Float16)f1.z; a[mi].h[7]=(_Float16)f1.w;
                a[mi].h[8]=(_Float16)f2.x; a[mi].h[9]=(_Float16)f2.y;
                a[mi].h[10]=(_Float16)f2.z; a[mi].h[11]=(_Float16)f2.w;
                a[mi].h[12]=(_Float16)f3.x; a[mi].h[13]=(_Float16)f3.y;
                a[mi].h[14]=(_Float16)f3.z; a[mi].h[15]=(_Float16)f3.w;
            }
        }
        // ---- B fragments (32x16 KxN): B[k][n] = W[col][k]; lane holds col,
        //      K runs [kb, kb+8) and [kb+16, kb+24), kb = half*8
        HFrag b[2];
        const int kb = k0 + half * 8;
        #pragma unroll
        for (int ni = 0; ni < 2; ++ni) {
            const size_t base = (size_t)(colTop0 + ni * 16 + l) * K + kb;
            const float4 f0 = ((const float4*)(W + base))[0];
            const float4 f1 = ((const float4*)(W + base))[1];
            const float4 f2 = ((const float4*)(W + base + 16))[0];
            const float4 f3 = ((const float4*)(W + base + 16))[1];
            b[ni].h[0]=(_Float16)f0.x; b[ni].h[1]=(_Float16)f0.y;
            b[ni].h[2]=(_Float16)f0.z; b[ni].h[3]=(_Float16)f0.w;
            b[ni].h[4]=(_Float16)f1.x; b[ni].h[5]=(_Float16)f1.y;
            b[ni].h[6]=(_Float16)f1.z; b[ni].h[7]=(_Float16)f1.w;
            b[ni].h[8]=(_Float16)f2.x; b[ni].h[9]=(_Float16)f2.y;
            b[ni].h[10]=(_Float16)f2.z; b[ni].h[11]=(_Float16)f2.w;
            b[ni].h[12]=(_Float16)f3.x; b[ni].h[13]=(_Float16)f3.y;
            b[ni].h[14]=(_Float16)f3.z; b[ni].h[15]=(_Float16)f3.w;
        }
        #pragma unroll
        for (int mi = 0; mi < 2; ++mi)
            #pragma unroll
            for (int ni = 0; ni < 2; ++ni)
                acc[mi][ni] = wmma16(a[mi], b[ni], acc[mi][ni]);
    }

    // ---- Store. C/D layout: N = l; VGPR r holds M = r + 8*half.
    #pragma unroll
    for (int mi = 0; mi < 2; ++mi) {
        #pragma unroll
        for (int ni = 0; ni < 2; ++ni) {
            const int col = colTop0 + ni * 16 + l;
            const float bv = bias[col];
            #pragma unroll
            for (int r = 0; r < 8; ++r) {
                const int m = rowTop0 + mi * 16 + r + half * 8;
                const float val = acc[mi][ni][r] + bv;
                const int h = col >> 6, d = col & 63;         // head, dim
                const int bi = m >> 11, s = m & (SEQ - 1);    // batch, seq
                if constexpr (MODE == OUT_HS) {
                    ((_Float16*)Yout)[(((size_t)bi * NHEADS + h) * SEQ + s) * DK + d] =
                        (_Float16)val;
                } else if constexpr (MODE == OUT_HS_T) {
                    ((_Float16*)Yout)[(((size_t)bi * NHEADS + h) * DK + d) * SEQ + s] =
                        (_Float16)val;
                } else {
                    ((float*)Yout)[(size_t)m * N + col] = val;
                }
            }
        }
    }
}

// ---------------------------------------------------------------------------
// Attention: per block = one (b*NH+h, 16-query-row strip).
// Phase 1: scores(16x2048) = Q(16x64) @ K^T / 8 -> LDS (f32)
// Phase 2: softmax in LDS, normalized P -> d_out (p_attn) and stays in LDS
// Phase 3: X(16x64) = P @ V via WMMA (V stored transposed -> contiguous
//          v8h B-fragment loads), cross-wave reduce in LDS, f16 -> xws
// Dynamic LDS: 16*2048*4 + 2*16*64*4 = 139264 B (fits 320 KB/WGP)
// ---------------------------------------------------------------------------
__global__ void __launch_bounds__(256)
attn_kernel(const _Float16* __restrict__ qws,
            const _Float16* __restrict__ kws,
            const _Float16* __restrict__ vws,   // transposed: [B*NH][DK][SEQ]
            float* __restrict__ pout,
            _Float16* __restrict__ xws)
{
    extern __shared__ char smem[];
    float* sc    = (float*)smem;                                      // [16][SEQ]
    float* xpart = (float*)(smem + (size_t)16 * SEQ * sizeof(float)); // [2][16][64]

    const int tid  = threadIdx.x;
    const int wave = tid >> 5;
    const int lane = tid & 31;
    const int half = lane >> 4;
    const int l    = lane & 15;

    const int bh    = blockIdx.y;          // b*NHEADS + h
    const int strip = blockIdx.x;          // 16-row query strip
    const _Float16* Q  = qws + (size_t)bh * SEQ * DK;
    const _Float16* Kp = kws + (size_t)bh * SEQ * DK;
    const _Float16* Vt = vws + (size_t)bh * DK * SEQ;  // [DK][SEQ]

    // ---- Phase 1: scores = Q K^T * (1/sqrt(64))
    HFrag aq[2];   // Q tile 16x64 as two 16x32 A fragments (k-steps)
    {
        const size_t rbase = (size_t)(strip * 16 + l) * DK;
        #pragma unroll
        for (int t = 0; t < 2; ++t) {
            const int kb1 = t * 32 + half * 8;
            aq[t].seg[0] = *(const v8h*)(Q + rbase + kb1);
            aq[t].seg[1] = *(const v8h*)(Q + rbase + kb1 + 16);
        }
    }
    // Warm the L2/L0 path for the V^T rows this wave will read in phase 3.
    __builtin_prefetch(Vt + (size_t)((wave & 3) * 16 + l) * SEQ + (wave >> 2) * (SEQ / 2), 0, 1);

    for (int ct = wave * 16; ct < SEQ; ct += 128) {
        v8f acc = {};
        const int col = ct + l;            // key index for this lane's B column
        #pragma unroll
        for (int t = 0; t < 2; ++t) {
            HFrag bk;                      // B[k][n] = K[col][k], k runs [8h,8h+8) and [16+8h,24+8h)
            bk.seg[0] = *(const v8h*)(Kp + (size_t)col * DK + t * 32 + half * 8);
            bk.seg[1] = *(const v8h*)(Kp + (size_t)col * DK + t * 32 + half * 8 + 16);
            acc = wmma16(aq[t], bk, acc);
        }
        #pragma unroll
        for (int r = 0; r < 8; ++r)
            sc[(r + half * 8) * SEQ + col] = acc[r] * 0.125f;
    }
    __syncthreads();

    // ---- Phase 2: softmax (wave w owns rows 2w, 2w+1); write p_attn
    #pragma unroll
    for (int rr2 = 0; rr2 < 2; ++rr2) {
        const int rr = wave * 2 + rr2;
        float* row = sc + (size_t)rr * SEQ;
        float mx = -3.4e38f;
        for (int j = lane; j < SEQ; j += 32) mx = fmaxf(mx, row[j]);
        #pragma unroll
        for (int off = 16; off > 0; off >>= 1) mx = fmaxf(mx, __shfl_xor(mx, off, 32));
        float sum = 0.f;
        for (int j = lane; j < SEQ; j += 32) { float e = __expf(row[j] - mx); row[j] = e; sum += e; }
        #pragma unroll
        for (int off = 16; off > 0; off >>= 1) sum += __shfl_xor(sum, off, 32);
        const float inv = 1.0f / sum;
        float* pg = pout + ((size_t)bh * SEQ + strip * 16 + rr) * SEQ;
        for (int j = lane; j < SEQ; j += 32) { float p = row[j] * inv; row[j] = p; pg[j] = p; }
    }
    __syncthreads();

    // ---- Phase 3: X = P @ V.  8 waves = 4 d-tiles x 2 K-halves.
    const int dtile = wave & 3;
    const int khalf = wave >> 2;
    const int dcol  = dtile * 16 + l;
    const _Float16* vrow = Vt + (size_t)dcol * SEQ;   // V^T row = fixed d, k-major
    v8f acc = {};
    for (int k0 = khalf * (SEQ / 2); k0 < (khalf + 1) * (SEQ / 2); k0 += 32) {
        HFrag ap, bv;
        const int kb1 = k0 + half * 8;
        #pragma unroll
        for (int e = 0; e < 8; ++e) ap.h[e]     = (_Float16)sc[(size_t)l * SEQ + kb1 + e];
        #pragma unroll
        for (int e = 0; e < 8; ++e) ap.h[8 + e] = (_Float16)sc[(size_t)l * SEQ + kb1 + 16 + e];
        // B[k][n] = V[k][dcol] = V^T[dcol][k]: two 16B vector loads (k runs kb1.., kb1+16..)
        bv.seg[0] = *(const v8h*)(vrow + kb1);
        bv.seg[1] = *(const v8h*)(vrow + kb1 + 16);
        acc = wmma16(ap, bv, acc);
    }
    #pragma unroll
    for (int r = 0; r < 8; ++r)
        xpart[(khalf * 16 + (r + half * 8)) * 64 + dcol] = acc[r];
    __syncthreads();
    if (khalf == 0) {
        const int bi = bh >> 4, hh = bh & 15;
        #pragma unroll
        for (int r = 0; r < 8; ++r) {
            const int m = r + half * 8;
            const float v = xpart[m * 64 + dcol] + xpart[(16 + m) * 64 + dcol];
            xws[((size_t)(bi * SEQ + strip * 16 + m)) * HIDDEN + hh * DK + dcol] = (_Float16)v;
        }
    }
}

// ---------------------------------------------------------------------------
extern "C" void kernel_launch(void* const* d_in, const int* in_sizes, int n_in,
                              void* d_out, int out_size, void* d_ws, size_t ws_size,
                              hipStream_t stream)
{
    (void)in_sizes; (void)n_in; (void)out_size; (void)ws_size;
    const float* query = (const float*)d_in[0];
    const float* key_  = (const float*)d_in[1];
    const float* value = (const float*)d_in[2];
    const float* Wq = (const float*)d_in[3];
    const float* bq = (const float*)d_in[4];
    const float* Wk = (const float*)d_in[5];
    const float* bk = (const float*)d_in[6];
    const float* Wv = (const float*)d_in[7];
    const float* bv = (const float*)d_in[8];
    const float* Wo = (const float*)d_in[9];
    const float* bo = (const float*)d_in[10];

    // f16 workspace: q, k (head-split), v (head-split transposed), x ([M,H])
    _Float16* qws = (_Float16*)d_ws;
    _Float16* kws = qws + (size_t)MROWS * HIDDEN;
    _Float16* vws = kws + (size_t)MROWS * HIDDEN;
    _Float16* xws = vws + (size_t)MROWS * HIDDEN;

    float* outp = (float*)d_out;                          // [B,S,H]
    float* pout = outp + (size_t)MROWS * HIDDEN;          // [B,NH,S,S]

    dim3 blk(256);
    dim3 ggrid(HIDDEN / 128, MROWS / 64);                 // (8, 64)

    hipLaunchKernelGGL((gemm_wmma_kernel<false, OUT_HS>), ggrid, blk, 0, stream,
                       (const void*)query, Wq, bq, (void*)qws);
    hipLaunchKernelGGL((gemm_wmma_kernel<false, OUT_HS>), ggrid, blk, 0, stream,
                       (const void*)key_, Wk, bk, (void*)kws);
    hipLaunchKernelGGL((gemm_wmma_kernel<false, OUT_HS_T>), ggrid, blk, 0, stream,
                       (const void*)value, Wv, bv, (void*)vws);

    dim3 agrid(SEQ / 16, BATCH * NHEADS);                 // (128, 32)
    size_t shmem = (size_t)16 * SEQ * sizeof(float) + (size_t)2 * 16 * 64 * sizeof(float);
    hipLaunchKernelGGL(attn_kernel, agrid, blk, shmem, stream, qws, kws, vws, pout, xws);

    hipLaunchKernelGGL((gemm_wmma_kernel<true, OUT_F32>), ggrid, blk, 0, stream,
                       (const void*)xws, Wo, bo, d_out);
}
